// HET_RelationalAttLayer_58866821759514
// MI455X (gfx1250) — matrix operand order repaired and checked by reading, not compiled, not run on hardware
//
#include <hip/hip_runtime.h>
#include <hip/hip_bf16.h>
#include <cstdint>
#include <cstddef>

#define GN_NODES 30000
#define GN_EDGES 480000
#define GIN_FEAT 256
#define GOUT_FEAT 256
#define GN_RELS 8
#define GN_HEADS 4
#define GHEAD_DIM 64
#define GLEAKY 0.2f

typedef __attribute__((ext_vector_type(16))) __bf16 v16bf;
typedef __attribute__((ext_vector_type(8)))  float  v8f;
typedef __attribute__((ext_vector_type(4)))  unsigned int v4u;

union Frag16 { v16bf bf; v4u u[2]; };

__device__ __forceinline__ unsigned short f2bf(float f) {
    unsigned int u = __float_as_uint(f);
    u += 0x7FFFu + ((u >> 16) & 1u);          // round-to-nearest-even
    return (unsigned short)(u >> 16);
}

__device__ __forceinline__ float atomicMaxFloat(float* addr, float val) {
    // signed/unsigned split trick: int ordering matches float for >=0,
    // reversed unsigned ordering for <0. Works with -inf init.
    if (val >= 0.0f)
        return __int_as_float(atomicMax((int*)addr, __float_as_int(val)));
    else
        return __uint_as_float(atomicMin((unsigned int*)addr, __float_as_uint(val)));
}

// ---------------- Stage A: fp32 -> bf16 conversions ----------------

__global__ void cvt_inputs_bf16(const float* __restrict__ x, unsigned short* __restrict__ xb) {
    int id = blockIdx.x * blockDim.x + threadIdx.x;
    if (id < GN_NODES * GIN_FEAT) xb[id] = f2bf(x[id]);
}

// conv_weights [R][H][IN][D] -> Wt [R][out_col=h*64+d][k]  (bf16, K contiguous)
__global__ void cvt_weights_bf16(const float* __restrict__ w, unsigned short* __restrict__ wt) {
    int id = blockIdx.x * blockDim.x + threadIdx.x;
    if (id >= GN_RELS * GOUT_FEAT * GIN_FEAT) return;
    int k   = id & (GIN_FEAT - 1);
    int col = (id >> 8) & (GOUT_FEAT - 1);
    int r   = id >> 16;
    int h = col >> 6, d = col & 63;
    float v = w[(((size_t)r * GN_HEADS + h) * GIN_FEAT + k) * GHEAD_DIM + d];
    wt[id] = f2bf(v);
}

// ---------------- Stage B: WMMA GEMM  feat[r] = Xb @ Wt[r]^T ----------------
// block = 256 threads = 8 waves; wave w computes rows m0+16w..+15, cols n0..n0+63.
// The 64x256 bf16 weight tile (32 KB, contiguous in Wt) is shared by all 8 waves:
// stage it once into LDS with gfx1250 async global->LDS copies (ASYNCcnt), then
// feed B fragments from LDS (ds_load_b128) while A streams from global (LOADcnt).
__global__ void gemm_wmma_bf16(const unsigned short* __restrict__ Xb,
                               const unsigned short* __restrict__ Wt,
                               float* __restrict__ feat) {
    __shared__ unsigned short ldsB[64 * GIN_FEAT];   // 32 KB: [colLocal][k]

    const int tid  = threadIdx.x;
    const int wave = tid >> 5;
    const int lane = tid & 31;
    const int half = lane >> 4;          // 0: lanes 0-15, 1: lanes 16-31
    const int lrow = lane & 15;
    const int r  = blockIdx.z;
    const int n0 = blockIdx.y * 64;
    const int m0 = blockIdx.x * 128 + wave * 16;

    // ---- async-stage B tile: 32 KB = 8 iters x 256 threads x 16 B ----
    {
        const char* gB = (const char*)(Wt + ((size_t)r * GOUT_FEAT + n0) * GIN_FEAT);
        // flat LDS pointers carry the wave-relative LDS byte offset in addr[31:0]
        unsigned ldsBase = (unsigned)(uintptr_t)(&ldsB[0]);
        #pragma unroll
        for (int i = 0; i < 8; ++i) {
            int byteOff = (i * 256 + tid) * 16;
            unsigned ldsAddr = ldsBase + (unsigned)byteOff;
            const void* gp = gB + byteOff;
            asm volatile("global_load_async_to_lds_b128 %0, %1, off"
                         :: "v"(ldsAddr), "v"(gp) : "memory");
        }
        asm volatile("s_wait_asynccnt 0x0" ::: "memory");
        __syncthreads();
    }

    int arow = m0 + lrow;
    if (arow >= GN_NODES) arow = GN_NODES - 1;   // clamp loads; stores guarded
    const unsigned short* Arow = Xb + (size_t)arow * GIN_FEAT;

    v8f acc[4] = {};

    #pragma unroll
    for (int k0 = 0; k0 < GIN_FEAT; k0 += 32) {
        Frag16 a;
        // 16-bit A 16x32 layout: elems 0-7 -> K = k0 + half*8 + e
        //                        elems 8-15 -> K = k0 + 16 + half*8 + (e-8)
        a.u[0] = *(const v4u*)(Arow + k0 + half * 8);
        a.u[1] = *(const v4u*)(Arow + k0 + 16 + half * 8);
        #pragma unroll
        for (int t = 0; t < 4; ++t) {
            const unsigned short* Bcol = &ldsB[(t * 16 + lrow) * GIN_FEAT];
            Frag16 b;
            // 16-bit B 32x16 layout: elems 0-7 -> K = k0 + half*8 + e, elems 8-15 -> K = k0 + 16 + half*8 + (e-8)
            b.u[0] = *(const v4u*)(Bcol + k0 + half * 8);
            b.u[1] = *(const v4u*)(Bcol + k0 + 16 + half * 8);
            acc[t] = __builtin_amdgcn_wmma_f32_16x16x32_bf16(
                false, a.bf, false, b.bf, (short)0, acc[t], false, false);
        }
    }

    #pragma unroll
    for (int t = 0; t < 4; ++t) {
        int col = n0 + t * 16 + lrow;
        #pragma unroll
        for (int j = 0; j < 8; ++j) {
            int mrow = m0 + j + half * 8;      // C/D layout: VGPR j -> M = j (+8 for half 1)
            if (mrow < GN_NODES)
                feat[((size_t)r * GN_NODES + mrow) * GOUT_FEAT + col] = acc[t][j];
        }
    }
}

// ---------------- Stage C: per-(rel,node) attention dots ----------------
// one wave per (r,n): el_all[r,n,h] = feat[r,n,h,:]·attn_l[r,h,:]  (same for er)
__global__ void node_dots(const float* __restrict__ feat,
                          const float* __restrict__ attn_l,
                          const float* __restrict__ attn_r,
                          float* __restrict__ el_all,
                          float* __restrict__ er_all) {
    int gw   = (blockIdx.x * blockDim.x + threadIdx.x) >> 5;
    int lane = threadIdx.x & 31;
    if (gw >= GN_RELS * GN_NODES) return;
    int r = gw / GN_NODES;
    int c0 = lane * 8;
    int h  = c0 >> 6;
    const float* row = feat + (size_t)gw * GOUT_FEAT + c0;
    const float* al  = attn_l + (r * GN_HEADS + h) * GHEAD_DIM + (c0 & 63);
    const float* ar  = attn_r + (r * GN_HEADS + h) * GHEAD_DIM + (c0 & 63);
    float4 f0 = *(const float4*)(row);     float4 f1 = *(const float4*)(row + 4);
    float4 a0 = *(const float4*)(al);      float4 a1 = *(const float4*)(al + 4);
    float4 b0 = *(const float4*)(ar);      float4 b1 = *(const float4*)(ar + 4);
    float sl = f0.x*a0.x + f0.y*a0.y + f0.z*a0.z + f0.w*a0.w
             + f1.x*a1.x + f1.y*a1.y + f1.z*a1.z + f1.w*a1.w;
    float sr = f0.x*b0.x + f0.y*b0.y + f0.z*b0.z + f0.w*b0.w
             + f1.x*b1.x + f1.y*b1.y + f1.z*b1.z + f1.w*b1.w;
    #pragma unroll
    for (int m = 1; m < 8; m <<= 1) {       // reduce 8-lane head groups
        sl += __shfl_xor(sl, m, 32);
        sr += __shfl_xor(sr, m, 32);
    }
    if ((lane & 7) == 0) {
        el_all[(size_t)gw * GN_HEADS + h] = sl;
        er_all[(size_t)gw * GN_HEADS + h] = sr;
    }
}

// ---------------- Stage D: init buffers ----------------

__global__ void init_out_bias(const float* __restrict__ bias, float* __restrict__ out) {
    int id = blockIdx.x * blockDim.x + threadIdx.x;
    if (id < GN_NODES * GOUT_FEAT) out[id] = bias[id & (GOUT_FEAT - 1)];
}

__global__ void init_max_denom(float* __restrict__ m, float* __restrict__ den) {
    int id = blockIdx.x * blockDim.x + threadIdx.x;
    if (id < GN_NODES * GN_HEADS) { m[id] = -__builtin_huge_valf(); den[id] = 0.0f; }
}

// ---------------- Stage E: edge logits + segment max ----------------

__global__ void edge_logits_max(const int* __restrict__ esrc, const int* __restrict__ edst,
                                const int* __restrict__ etyp,
                                const float* __restrict__ el_all, const float* __restrict__ er_all,
                                float* __restrict__ ebuf, float* __restrict__ m) {
    int t = blockIdx.x * blockDim.x + threadIdx.x;
    if (t >= GN_EDGES * GN_HEADS) return;
    int e = t >> 2, h = t & 3;
    int rt = etyp[e], s = esrc[e], d = edst[e];
    float v = el_all[((size_t)rt * GN_NODES + s) * GN_HEADS + h]
            + er_all[((size_t)rt * GN_NODES + d) * GN_HEADS + h];
    v = (v > 0.0f) ? v : GLEAKY * v;        // leaky relu
    ebuf[t] = v;
    atomicMaxFloat(&m[d * GN_HEADS + h], v);
}

// ---------------- Stage F: exp + segment sum ----------------

__global__ void edge_exp_sum(const int* __restrict__ edst,
                             const float* __restrict__ m,
                             float* __restrict__ ebuf, float* __restrict__ den) {
    int t = blockIdx.x * blockDim.x + threadIdx.x;
    if (t >= GN_EDGES * GN_HEADS) return;
    int e = t >> 2, h = t & 3;
    int d = edst[e];
    float ex = __expf(ebuf[t] - m[d * GN_HEADS + h]);
    ebuf[t] = ex;
    atomicAdd(&den[d * GN_HEADS + h], ex);
}

// ---------------- Stage G: weighted aggregation ----------------
// one wave per edge; lane handles 8 contiguous output channels

__global__ void edge_aggregate(const int* __restrict__ esrc, const int* __restrict__ edst,
                               const int* __restrict__ etyp,
                               const float* __restrict__ feat, const float* __restrict__ ebuf,
                               const float* __restrict__ den, float* __restrict__ out) {
    int gw   = (blockIdx.x * blockDim.x + threadIdx.x) >> 5;
    int lane = threadIdx.x & 31;
    if (gw >= GN_EDGES) return;
    int rt = etyp[gw], s = esrc[gw], d = edst[gw];
    int c0 = lane * 8;
    int h  = c0 >> 6;
    float alpha = ebuf[(size_t)gw * GN_HEADS + h] / den[d * GN_HEADS + h];
    const float* fs = feat + ((size_t)rt * GN_NODES + s) * GOUT_FEAT + c0;
    float4 f0 = *(const float4*)(fs);
    float4 f1 = *(const float4*)(fs + 4);
    float* op = out + (size_t)d * GOUT_FEAT + c0;
    atomicAdd(op + 0, alpha * f0.x);
    atomicAdd(op + 1, alpha * f0.y);
    atomicAdd(op + 2, alpha * f0.z);
    atomicAdd(op + 3, alpha * f0.w);
    atomicAdd(op + 4, alpha * f1.x);
    atomicAdd(op + 5, alpha * f1.y);
    atomicAdd(op + 6, alpha * f1.z);
    atomicAdd(op + 7, alpha * f1.w);
}

// ---------------- launch ----------------

extern "C" void kernel_launch(void* const* d_in, const int* in_sizes, int n_in,
                              void* d_out, int out_size, void* d_ws, size_t ws_size,
                              hipStream_t stream) {
    const float* inputs  = (const float*)d_in[0];
    const float* weights = (const float*)d_in[1];
    const float* attn_l  = (const float*)d_in[2];
    const float* attn_r  = (const float*)d_in[3];
    const float* h_bias  = (const float*)d_in[4];
    const int*   esrc    = (const int*)d_in[5];
    const int*   edst    = (const int*)d_in[6];
    const int*   etyp    = (const int*)d_in[7];
    float* out = (float*)d_out;

    char* ws = (char*)d_ws;
    const size_t OFF_XB   = 0;                                  // bf16 inputs  15,360,000 B
    const size_t OFF_WT   = OFF_XB + 15360000;                  // bf16 weightsT 1,048,576 B
    const size_t OFF_FEAT = OFF_WT + 1048576;                   // f32 feat_all 245,760,000 B
    const size_t OFF_EL   = OFF_FEAT + 245760000;               // f32 el_all    3,840,000 B
    const size_t OFF_ER   = OFF_EL + 3840000;                   // f32 er_all    3,840,000 B
    const size_t OFF_E    = OFF_ER + 3840000;                   // f32 edge exp  7,680,000 B
    const size_t OFF_M    = OFF_E + 7680000;                    // f32 max         480,000 B
    const size_t OFF_D    = OFF_M + 480000;                     // f32 denom       480,000 B

    unsigned short* Xb   = (unsigned short*)(ws + OFF_XB);
    unsigned short* Wt   = (unsigned short*)(ws + OFF_WT);
    float* feat   = (float*)(ws + OFF_FEAT);
    float* el_all = (float*)(ws + OFF_EL);
    float* er_all = (float*)(ws + OFF_ER);
    float* ebuf   = (float*)(ws + OFF_E);
    float* mbuf   = (float*)(ws + OFF_M);
    float* den    = (float*)(ws + OFF_D);

    const int B = 256;

    cvt_inputs_bf16<<<(GN_NODES * GIN_FEAT + B - 1) / B, B, 0, stream>>>(inputs, Xb);
    cvt_weights_bf16<<<(GN_RELS * GOUT_FEAT * GIN_FEAT + B - 1) / B, B, 0, stream>>>(weights, Wt);

    dim3 gg((GN_NODES + 127) / 128, GOUT_FEAT / 64, GN_RELS);
    gemm_wmma_bf16<<<gg, B, 0, stream>>>(Xb, Wt, feat);

    node_dots<<<(GN_RELS * GN_NODES * 32 + B - 1) / B, B, 0, stream>>>(feat, attn_l, attn_r, el_all, er_all);

    init_out_bias<<<(GN_NODES * GOUT_FEAT + B - 1) / B, B, 0, stream>>>(h_bias, out);
    init_max_denom<<<(GN_NODES * GN_HEADS + B - 1) / B, B, 0, stream>>>(mbuf, den);

    edge_logits_max<<<(GN_EDGES * GN_HEADS + B - 1) / B, B, 0, stream>>>(esrc, edst, etyp, el_all, er_all, ebuf, mbuf);
    edge_exp_sum<<<(GN_EDGES * GN_HEADS + B - 1) / B, B, 0, stream>>>(edst, mbuf, ebuf, den);
    edge_aggregate<<<((size_t)GN_EDGES * 32 + B - 1) / B, B, 0, stream>>>(esrc, edst, etyp, feat, ebuf, den, out);
}
